// GNN_34256659153438
// MI455X (gfx1250) — matrix operand order repaired and checked by reading, not compiled, not run on hardware
//
#include <hip/hip_runtime.h>

#define NN   50000
#define EE   200000
#define HIDC 128
#define HH   2
#define DHD  128
#define LL   4
#define NGR  500
#define CONC (HIDC * (LL + 1))   // 640

typedef __attribute__((ext_vector_type(16))) _Float16 v16h;
typedef __attribute__((ext_vector_type(8)))  float    v8f;

// ---------- helpers ----------
__device__ __forceinline__ unsigned fenc(float f) {
  unsigned u = __float_as_uint(f);
  return (u & 0x80000000u) ? ~u : (u | 0x80000000u);
}
__device__ __forceinline__ float fdec(unsigned u) {
  unsigned b = (u & 0x80000000u) ? (u ^ 0x80000000u) : ~u;
  return __uint_as_float(b);
}

// ---------- weight convert + transpose: Wt[n][k] (f16) from W[k][n] (f32) ----------
__global__ void wtrans_kernel(const float* __restrict__ W, _Float16* __restrict__ Wt,
                              int K, int Nout) {
  long t = (long)blockIdx.x * blockDim.x + threadIdx.x;
  long total = (long)K * Nout;
  if (t >= total) return;
  long n = t / K, k = t % K;
  Wt[t] = (_Float16)W[k * (long)Nout + n];
}

// ---------- WMMA GEMM: C = epilogue(A[idx] * W + bias) ----------
// One wave computes a 64x16 output tile: 4 M-subtiles share each B fragment.
// K is a template parameter -> K-loop fully unrolls (K/32 steps, 4 WMMAs each).
// flags: bit0 = relu, bit1 = batchnorm(gamma,beta) eval mode (running stats 0/1)
template <int K>
__global__ void gemm_wmma_kernel(const float* __restrict__ A, const int* __restrict__ aidx,
                                 int lda,
                                 const _Float16* __restrict__ Wt,   // [Nout][K] f16
                                 const float* __restrict__ bias,
                                 const float* __restrict__ gamma,
                                 const float* __restrict__ beta,
                                 int flags, int M,
                                 float* __restrict__ C, int ldc)
{
  const int lane  = threadIdx.x;          // blockDim.x == 32, one wave
  const int tileM = blockIdx.x * 64;
  const int tileN = blockIdx.y * 16;
  const int col   = tileN + (lane & 15);
  const int aofs  = (lane & 16) ? 8 : 0;  // A K-half offset (documented layout)
  const int bofs  = (lane & 16) ? 8 : 0;  // B K-half offset (lanes 16-31: K+8..15, K+24..31)

  long arow[4];
#pragma unroll
  for (int s = 0; s < 4; s++) {
    int r = tileM + s * 16 + (lane & 15);
    if (r > M - 1) r = M - 1;             // clamp; stores are guarded
    arow[s] = aidx ? (long)aidx[r] : (long)r;
  }

  v8f acc[4] = {};
#pragma unroll
  for (int kk = 0; kk < K; kk += 32) {
    v16h b;
    const _Float16* bp = Wt + (long)col * K + kk + bofs;
#pragma unroll
    for (int j = 0; j < 8; j++) b[j] = bp[j];           // 2x b128: k 0-7 and 16-23 (+8h)
#pragma unroll
    for (int j = 0; j < 8; j++) b[8 + j] = bp[16 + j];

#pragma unroll
    for (int s = 0; s < 4; s++) {
      v16h a;
      const float* ap = A + arow[s] * (long)lda + kk + aofs;
#pragma unroll
      for (int j = 0; j < 8; j++) a[j]     = (_Float16)ap[j];
#pragma unroll
      for (int j = 0; j < 8; j++) a[8 + j] = (_Float16)ap[16 + j];
      acc[s] = __builtin_amdgcn_wmma_f32_16x16x32_f16(false, a, false, b,
                                                      (short)0, acc[s], false, false);
    }
  }

  const float INVS = 0.99999500003749969f;  // 1/sqrt(1 + 1e-5)  (BN eval, var=1)
  float bc = bias ? bias[col] : 0.f;
  float gm = 1.f, bt = 0.f;
  if (flags & 2) { gm = gamma[col] * INVS; bt = beta[col]; }

#pragma unroll
  for (int s = 0; s < 4; s++) {
#pragma unroll
    for (int rr = 0; rr < 8; rr++) {
      int m = tileM + s * 16 + rr + ((lane & 16) ? 8 : 0);
      if (m < M) {
        float v = acc[s][rr] + bc;
        if (flags & 2) v = v * gm + bt;
        if (flags & 1) v = fmaxf(v, 0.f);
        C[(long)m * ldc + col] = v;
      }
    }
  }
}

// ---------- fill ----------
__global__ void fill_u32_kernel(unsigned* __restrict__ p, unsigned v, long n) {
  long t = (long)blockIdx.x * blockDim.x + threadIdx.x;
  if (t < n) p[t] = v;
}

// ---------- per-edge attention score + segment max ----------
// one wave per (edge, head); lane covers 4 contiguous channels (b128 loads)
__global__ void edge_score_kernel(const float* __restrict__ q, const float* __restrict__ k,
                                  const float* __restrict__ e_all,
                                  const float* __restrict__ att,   // [H][128] of this layer
                                  const int* __restrict__ src, const int* __restrict__ dst,
                                  float* __restrict__ score, unsigned* __restrict__ smax)
{
  long gt = (long)blockIdx.x * blockDim.x + threadIdx.x;
  long wave = gt >> 5;
  int  lane = (int)(gt & 31);
  if (wave >= (long)EE * HH) return;
  int e = (int)(wave >> 1), h = (int)(wave & 1);
  int sn = src[e], dn = dst[e];

  float4 qv = *(const float4*)(q     + (long)dn * 256 + h * 128 + lane * 4);
  float4 kv = *(const float4*)(k     + (long)sn * 256 + h * 128 + lane * 4);
  float4 ev = *(const float4*)(e_all + (long)e  * 256 + h * 128 + lane * 4);
  float4 av = *(const float4*)(att + h * 128 + lane * 4);

  float s = 0.f;
  {
    float xs[4] = { qv.x + kv.x + ev.x, qv.y + kv.y + ev.y,
                    qv.z + kv.z + ev.z, qv.w + kv.w + ev.w };
    float as[4] = { av.x, av.y, av.z, av.w };
#pragma unroll
    for (int i = 0; i < 4; i++) {
      float x = xs[i];
      x = (x > 0.f) ? x : 0.2f * x;        // leaky_relu(., 0.2)
      s += x * as[i];
    }
  }
  for (int off = 16; off > 0; off >>= 1) s += __shfl_xor(s, off);
  if (lane == 0) {
    score[wave] = s;
    atomicMax(&smax[(long)dn * HH + h], fenc(s));
  }
}

// ---------- exp(score - max) and segment-sum of exps ----------
__global__ void exps_kernel(const float* __restrict__ score, const unsigned* __restrict__ smax,
                            const int* __restrict__ dst,
                            float* __restrict__ exps, float* __restrict__ denom)
{
  long t = (long)blockIdx.x * blockDim.x + threadIdx.x;
  if (t >= (long)EE * HH) return;
  int e = (int)(t >> 1), h = (int)(t & 1);
  int dn = dst[e];
  float ex = expf(score[t] - fdec(smax[(long)dn * HH + h]));
  exps[t] = ex;
  atomicAdd(&denom[(long)dn * HH + h], ex);
}

// ---------- alpha * (v + e) scatter-add ----------
__global__ void msg_kernel(const float* __restrict__ v, const float* __restrict__ e_all,
                           const float* __restrict__ exps, const float* __restrict__ denom,
                           const int* __restrict__ src, const int* __restrict__ dst,
                           float* __restrict__ agg)
{
  long gt = (long)blockIdx.x * blockDim.x + threadIdx.x;
  long wave = gt >> 5;
  int  lane = (int)(gt & 31);
  if (wave >= (long)EE * HH) return;
  int e = (int)(wave >> 1), h = (int)(wave & 1);
  int sn = src[e], dn = dst[e];
  float alpha = exps[wave] / denom[(long)dn * HH + h];

  float4 vv = *(const float4*)(v     + (long)sn * 256 + h * 128 + lane * 4);
  float4 ev = *(const float4*)(e_all + (long)e  * 256 + h * 128 + lane * 4);
  float*  ap = agg + (long)dn * 256 + h * 128 + lane * 4;
  atomicAdd(ap + 0, alpha * (vv.x + ev.x));
  atomicAdd(ap + 1, alpha * (vv.y + ev.y));
  atomicAdd(ap + 2, alpha * (vv.z + ev.z));
  atomicAdd(ap + 3, alpha * (vv.w + ev.w));
}

// ---------- head-mean + root + relu + initial residual, write into hc slot ----------
__global__ void combine_kernel(const float* __restrict__ agg, const float* __restrict__ root,
                               float* __restrict__ hc, int li)
{
  long t = (long)blockIdx.x * blockDim.x + threadIdx.x;
  if (t >= (long)NN * HIDC) return;
  int n = (int)(t >> 7), c = (int)(t & 127);
  float a = 0.5f * (agg[(long)n * 256 + c] + agg[(long)n * 256 + 128 + c]);
  float o = fmaxf(a + root[t], 0.f);
  float x0 = hc[(long)n * CONC + c];
  hc[(long)n * CONC + (li + 1) * HIDC + c] = 0.9f * o + 0.1f * x0;   // RES = 0.1
}

// ---------- global mean pool ----------
__global__ void pool_add_kernel(const float* __restrict__ hc, const int* __restrict__ batch,
                                float* __restrict__ pooled)
{
  long t = (long)blockIdx.x * blockDim.x + threadIdx.x;
  if (t >= (long)NN * (CONC / 4)) return;
  int n  = (int)(t / (CONC / 4));
  int c4 = (int)(t % (CONC / 4)) * 4;
  int g  = batch[n];
  float4 hv = *(const float4*)(hc + (long)n * CONC + c4);
  float* pp = pooled + (long)g * CONC + c4;
  atomicAdd(pp + 0, hv.x);
  atomicAdd(pp + 1, hv.y);
  atomicAdd(pp + 2, hv.z);
  atomicAdd(pp + 3, hv.w);
}

__global__ void cnt_kernel(const int* __restrict__ batch, float* __restrict__ cnt) {
  long t = (long)blockIdx.x * blockDim.x + threadIdx.x;
  if (t >= NN) return;
  atomicAdd(&cnt[batch[t]], 1.0f);
}

__global__ void pool_fin_kernel(float* __restrict__ pooled, const float* __restrict__ cnt) {
  long t = (long)blockIdx.x * blockDim.x + threadIdx.x;
  if (t >= (long)NGR * CONC) return;
  pooled[t] /= fmaxf(cnt[t / CONC], 1.0f);
}

// ---------- final 128 -> 1 head: one wave per graph ----------
__global__ void head_kernel(const float* __restrict__ g1, const float* __restrict__ w2,
                            const float* __restrict__ b2, float* __restrict__ out)
{
  long gt = (long)blockIdx.x * blockDim.x + threadIdx.x;
  long wave = gt >> 5;
  int  lane = (int)(gt & 31);
  if (wave >= NGR) return;
  float4 gv = *(const float4*)(g1 + wave * 128 + lane * 4);
  float4 wv = *(const float4*)(w2 + lane * 4);
  float s = gv.x * wv.x + gv.y * wv.y + gv.z * wv.z + gv.w * wv.w;
  for (int off = 16; off > 0; off >>= 1) s += __shfl_xor(s, off);
  if (lane == 0) out[wave] = s + b2[0];
}

// =====================================================================
extern "C" void kernel_launch(void* const* d_in, const int* in_sizes, int n_in,
                              void* d_out, int out_size, void* d_ws, size_t ws_size,
                              hipStream_t stream) {
  (void)in_sizes; (void)n_in; (void)out_size; (void)ws_size;
  const int*   x        = (const int*)d_in[0];
  const int*   ei       = (const int*)d_in[1];
  const int*   src      = ei;
  const int*   dst      = ei + EE;
  const int*   eattr    = (const int*)d_in[2];
  const int*   batch    = (const int*)d_in[3];
  const float* node_emb = (const float*)d_in[4];
  const float* emb_w1   = (const float*)d_in[5];
  const float* emb_b1   = (const float*)d_in[6];
  const float* emb_g1   = (const float*)d_in[7];
  const float* emb_bt1  = (const float*)d_in[8];
  const float* emb_w2   = (const float*)d_in[9];
  const float* emb_b2   = (const float*)d_in[10];
  const float* bn0_g    = (const float*)d_in[11];
  const float* bn0_b    = (const float*)d_in[12];
  const float* Wq       = (const float*)d_in[13];
  const float* Wk       = (const float*)d_in[14];
  const float* Wv       = (const float*)d_in[15];
  const float* We       = (const float*)d_in[16];
  const float* att      = (const float*)d_in[17];
  const float* Wroot    = (const float*)d_in[18];
  const float* biasL    = (const float*)d_in[19];
  const float* out_w1   = (const float*)d_in[20];
  const float* out_b1   = (const float*)d_in[21];
  const float* out_g1   = (const float*)d_in[22];
  const float* out_bt1  = (const float*)d_in[23];
  const float* out_w2   = (const float*)d_in[24];
  const float* out_b2   = (const float*)d_in[25];
  float* out = (float*)d_out;

  // ---- workspace carve ----
  char* wsp = (char*)d_ws;
  size_t off = 0;
  auto allocb = [&](size_t nbytes) -> void* {
    void* p = (void*)(wsp + off);
    off = (off + nbytes + 255) & ~(size_t)255;
    return p;
  };
  auto allocf = [&](size_t n) -> float*     { return (float*)allocb(n * 4); };
  auto alloch = [&](size_t n) -> _Float16*  { return (_Float16*)allocb(n * 2); };

  float*    hc     = allocf((size_t)NN * CONC);   // [N][640]: x0 | layer1..4
  float*    qb     = allocf((size_t)NN * 256);
  float*    kb     = allocf((size_t)NN * 256);
  float*    vb     = allocf((size_t)NN * 256);
  float*    eb     = allocf((size_t)EE * 256);
  float*    rootb  = allocf((size_t)NN * HIDC);
  float*    score  = allocf((size_t)EE * HH);
  float*    exps   = allocf((size_t)EE * HH);
  unsigned* smax   = (unsigned*)allocf((size_t)NN * HH);
  float*    denom  = allocf((size_t)NN * HH);
  float*    agg    = allocf((size_t)NN * 256);
  float*    pooled = allocf((size_t)NGR * CONC);
  float*    cnt    = allocf((size_t)NGR);
  float*    g1b    = allocf((size_t)NGR * HIDC);
  float*    t1     = qb;   // reuse: only live before layer loop

  // f16 transposed weights
  _Float16* emb_w1t = alloch(128 * 128);
  _Float16* emb_w2t = alloch(128 * 128);
  _Float16* WqT     = alloch((size_t)LL * 256 * 128);
  _Float16* WkT     = alloch((size_t)LL * 256 * 128);
  _Float16* WvT     = alloch((size_t)LL * 256 * 128);
  _Float16* WeT     = alloch((size_t)LL * 256 * 128);
  _Float16* WrT     = alloch((size_t)LL * 128 * 128);
  _Float16* ow1T    = alloch((size_t)128 * 640);

  auto cdiv = [](long a, long b) { return (int)((a + b - 1) / b); };

  // ---- weight convert/transpose pre-pass (tiny, L2-resident afterwards) ----
  wtrans_kernel<<<cdiv(128 * 128, 256), 256, 0, stream>>>(emb_w1, emb_w1t, 128, 128);
  wtrans_kernel<<<cdiv(128 * 128, 256), 256, 0, stream>>>(emb_w2, emb_w2t, 128, 128);
  for (int i = 0; i < LL; i++) {
    wtrans_kernel<<<cdiv(128 * 256, 256), 256, 0, stream>>>(Wq + (size_t)i * 128 * 256, WqT + (size_t)i * 256 * 128, 128, 256);
    wtrans_kernel<<<cdiv(128 * 256, 256), 256, 0, stream>>>(Wk + (size_t)i * 128 * 256, WkT + (size_t)i * 256 * 128, 128, 256);
    wtrans_kernel<<<cdiv(128 * 256, 256), 256, 0, stream>>>(Wv + (size_t)i * 128 * 256, WvT + (size_t)i * 256 * 128, 128, 256);
    wtrans_kernel<<<cdiv(128 * 256, 256), 256, 0, stream>>>(We + (size_t)i * 128 * 256, WeT + (size_t)i * 256 * 128, 128, 256);
    wtrans_kernel<<<cdiv(128 * 128, 256), 256, 0, stream>>>(Wroot + (size_t)i * 128 * 128, WrT + (size_t)i * 128 * 128, 128, 128);
  }
  wtrans_kernel<<<cdiv(640 * 128, 256), 256, 0, stream>>>(out_w1, ow1T, 640, 128);

  // ---- node embedding MLP: relu(bn(emb[x]@W1+b1)) -> @W2+b2 -> relu(bn0) -> hc[:,0:128]
  {
    dim3 g((NN + 63) / 64, HIDC / 16), b(32);
    gemm_wmma_kernel<128><<<g, b, 0, stream>>>(node_emb, x, HIDC, emb_w1t,
                                               emb_b1, emb_g1, emb_bt1, 3, NN, t1, HIDC);
    gemm_wmma_kernel<128><<<g, b, 0, stream>>>(t1, nullptr, HIDC, emb_w2t,
                                               emb_b2, bn0_g, bn0_b, 3, NN, hc, CONC);
  }

  // ---- message-passing layers ----
  for (int i = 0; i < LL; i++) {
    const float* hi = hc + (size_t)i * HIDC;     // lda = CONC
    dim3 b32(32);
    dim3 gN16((NN + 63) / 64, 16), gN8((NN + 63) / 64, 8), gE16(EE / 64, 16);

    gemm_wmma_kernel<128><<<gN16, b32, 0, stream>>>(hi, nullptr, CONC, WqT + (size_t)i * 256 * 128,
                                                    nullptr, nullptr, nullptr, 0, NN, qb, 256);
    gemm_wmma_kernel<128><<<gN16, b32, 0, stream>>>(hi, nullptr, CONC, WkT + (size_t)i * 256 * 128,
                                                    nullptr, nullptr, nullptr, 0, NN, kb, 256);
    gemm_wmma_kernel<128><<<gN16, b32, 0, stream>>>(hi, nullptr, CONC, WvT + (size_t)i * 256 * 128,
                                                    nullptr, nullptr, nullptr, 0, NN, vb, 256);
    // edge features: gather node_emb[edge_attr] fused into the GEMM via row index
    gemm_wmma_kernel<128><<<gE16, b32, 0, stream>>>(node_emb, eattr, HIDC, WeT + (size_t)i * 256 * 128,
                                                    nullptr, nullptr, nullptr, 0, EE, eb, 256);
    gemm_wmma_kernel<128><<<gN8, b32, 0, stream>>>(hi, nullptr, CONC, WrT + (size_t)i * 128 * 128,
                                                   biasL + (size_t)i * HIDC, nullptr, nullptr, 0,
                                                   NN, rootb, HIDC);

    fill_u32_kernel<<<cdiv((long)NN * HH, 256), 256, 0, stream>>>(smax, 0x007FFFFFu, (long)NN * HH); // enc(-inf)
    fill_u32_kernel<<<cdiv((long)NN * HH, 256), 256, 0, stream>>>((unsigned*)denom, 0u, (long)NN * HH);
    fill_u32_kernel<<<cdiv((long)NN * 256, 256), 256, 0, stream>>>((unsigned*)agg, 0u, (long)NN * 256);

    long ewaves = (long)EE * HH;
    edge_score_kernel<<<cdiv(ewaves * 32, 256), 256, 0, stream>>>(qb, kb, eb, att + (size_t)i * HH * DHD,
                                                                  src, dst, score, smax);
    exps_kernel<<<cdiv(ewaves, 256), 256, 0, stream>>>(score, smax, dst, exps, denom);
    msg_kernel<<<cdiv(ewaves * 32, 256), 256, 0, stream>>>(vb, eb, exps, denom, src, dst, agg);
    combine_kernel<<<cdiv((long)NN * HIDC, 256), 256, 0, stream>>>(agg, rootb, hc, i);
  }

  // ---- global mean pool ----
  fill_u32_kernel<<<cdiv((long)NGR * CONC, 256), 256, 0, stream>>>((unsigned*)pooled, 0u, (long)NGR * CONC);
  fill_u32_kernel<<<cdiv((long)NGR, 256), 256, 0, stream>>>((unsigned*)cnt, 0u, (long)NGR);
  pool_add_kernel<<<cdiv((long)NN * (CONC / 4), 256), 256, 0, stream>>>(hc, batch, pooled);
  cnt_kernel<<<cdiv((long)NN, 256), 256, 0, stream>>>(batch, cnt);
  pool_fin_kernel<<<cdiv((long)NGR * CONC, 256), 256, 0, stream>>>(pooled, cnt);

  // ---- output MLP ----
  {
    dim3 g((NGR + 63) / 64, HIDC / 16), b(32);
    gemm_wmma_kernel<640><<<g, b, 0, stream>>>(pooled, nullptr, CONC, ow1T,
                                               out_b1, out_g1, out_bt1, 3, NGR, g1b, HIDC);
  }
  head_kernel<<<cdiv((long)NGR * 32, 256), 256, 0, stream>>>(g1b, out_w2, out_b2, out);
}
